// MultiHeadAttention_23201413333084
// MI455X (gfx1250) — matrix operand order repaired and checked by reading, not compiled, not run on hardware
//
#include <hip/hip_runtime.h>
#include <hip/hip_bf16.h>

typedef __attribute__((ext_vector_type(16))) __bf16       v16bf;
typedef __attribute__((ext_vector_type(8)))  float        v8f;
typedef __attribute__((ext_vector_type(4)))  int          v4i_;
typedef __attribute__((ext_vector_type(8)))  int          v8i_;
typedef __attribute__((ext_vector_type(4)))  unsigned int v4u_;

#define AS1 __attribute__((address_space(1)))
#define AS3 __attribute__((address_space(3)))

#if defined(__gfx1250__) && __has_builtin(__builtin_amdgcn_global_load_async_to_lds_b128)
#define HAVE_ASYNC_LDS 1
#else
#define HAVE_ASYNC_LDS 0
#endif

#if defined(__gfx1250__) && __has_builtin(__builtin_amdgcn_tensor_load_to_lds) && \
    __has_builtin(__builtin_amdgcn_s_wait_tensorcnt)
#define HAVE_TDM 1
#else
#define HAVE_TDM 0
#endif

__device__ __forceinline__ void async_b128(const void* g, void* l) {
#if HAVE_ASYNC_LDS
    __builtin_amdgcn_global_load_async_to_lds_b128(
        (AS1 v4i_*)(void*)g, (AS3 v4i_*)l, 0, 0);
#else
    *reinterpret_cast<uint4*>(l) = *reinterpret_cast<const uint4*>(g);
#endif
}

__device__ __forceinline__ void async_wait_all() {
#if HAVE_ASYNC_LDS
#if __has_builtin(__builtin_amdgcn_s_wait_asynccnt)
    __builtin_amdgcn_s_wait_asynccnt(0);
#else
    asm volatile("s_wait_asynccnt 0" ::: "memory");
#endif
#endif
}

__device__ __forceinline__ unsigned int lds_offset_of(const void* p) {
    return (unsigned int)(uintptr_t)(AS3 const char*)p;
}

#if HAVE_TDM
// 2D bf16 tile DMA: tile_d1 rows of tile_d0 elements, global row stride row_stride
// (elements), optional LDS padding (pad codes per D# spec; pad_amt_code < 0 disables).
__device__ __forceinline__ void tdm_load_2d_bf16(const void* gaddr, unsigned int lds_off,
                                                 unsigned int tile_d0, unsigned int tile_d1,
                                                 unsigned int tensor_d1, unsigned int row_stride,
                                                 int pad_int_code, int pad_amt_code) {
    unsigned long long ga = (unsigned long long)(uintptr_t)gaddr;
    v4u_ g0 = { 1u,                                    // count = 1 (valid descriptor)
                lds_off,                               // lds_addr  [63:32]
                (unsigned int)ga,                      // global_addr[31:0]
                (unsigned int)((ga >> 32) & 0x01FFFFFFu) | (2u << 30) }; // addr[56:32] | type=2
    unsigned int w0 = (1u << 16);                      // data_size = 2 bytes
    if (pad_amt_code >= 0)
        w0 |= (1u << 20) | ((unsigned int)pad_int_code << 22) |
              ((unsigned int)pad_amt_code << 25);
    unsigned int td0 = tile_d0;                        // tensor_dim0 == tile width
    v8i_ g1 = { (int)w0,
                (int)((td0 & 0xFFFFu) << 16),                               // dim0[15:0]
                (int)((td0 >> 16) | ((tensor_d1 & 0xFFFFu) << 16)),         // dim0[31:16]|dim1[15:0]
                (int)((tensor_d1 >> 16) | ((tile_d0 & 0xFFFFu) << 16)),     // dim1[31:16]|tile0
                (int)(tile_d1 & 0xFFFFu),                                   // tile1 (tile2=0)
                (int)row_stride,                                            // dim0_stride[31:0]
                0, 0 };
    v4i_ gz4 = {0, 0, 0, 0};
    v8i_ gz8 = {0, 0, 0, 0, 0, 0, 0, 0};
    __builtin_amdgcn_tensor_load_to_lds(g0, g1, gz4, gz4, gz8, 0);
}
#endif

#define Bc 4
#define Tc 2048
#define Dc 1024
#define Hc 16
#define DHc 64
#define NEGINF (-1.0e30f)

// ---------------------------------------------------------------- converts
__global__ __launch_bounds__(256) void f32_to_bf16_kernel(const float* __restrict__ src,
                                                          __bf16* __restrict__ dst, size_t n) {
    size_t i = (size_t)blockIdx.x * blockDim.x + threadIdx.x;
    if (i < n) dst[i] = (__bf16)src[i];
}

__global__ __launch_bounds__(256) void rope_tables_kernel(float* __restrict__ cosb,
                                                          float* __restrict__ sinb) {
    int i = blockIdx.x * blockDim.x + threadIdx.x; // Tc*32
    if (i >= Tc * 32) return;
    int t = i >> 5, j = i & 31;
    float inv = __powf(10000.0f, -(float)(2 * j) / (float)DHc);
    float f = (float)t * inv;
    cosb[i] = __cosf(f);
    sinb[i] = __sinf(f);
}

// Apply RoPE in-place to q and k slices of fused qkv [B*T, 3*Dc]
__global__ __launch_bounds__(256) void rope_apply_kernel(__bf16* __restrict__ qkv,
                                                         const float* __restrict__ cosb,
                                                         const float* __restrict__ sinb) {
    size_t i = (size_t)blockIdx.x * blockDim.x + threadIdx.x;
    if (i >= (size_t)Bc * Tc * 2 * Hc * 32) return;
    int j = (int)(i & 31); i >>= 5;
    int h = (int)(i & 15); i >>= 4;
    int part = (int)(i & 1); i >>= 1;   // 0 = q, 1 = k
    int t = (int)(i % Tc);
    int b = (int)(i / Tc);
    size_t base = ((size_t)(b * Tc + t)) * (3 * Dc) + part * Dc + h * DHc + 2 * j;
    float x1 = (float)qkv[base], x2 = (float)qkv[base + 1];
    float c = cosb[t * 32 + j], s = sinb[t * 32 + j];
    qkv[base]     = (__bf16)(x1 * c - x2 * s);
    qkv[base + 1] = (__bf16)(x1 * s + x2 * c);
}

// ---------------------------------------------------------------- GEMM: C[M,N] = A[M,K] * W[N,K]^T
#define BM 128
#define BN 128
#define BK 32
#define TSTR 40   // 32 data + 8 pad (bf16) per LDS row

template <bool F32OUT>
__global__ __launch_bounds__(256) void gemm_bf16_wmma(const __bf16* __restrict__ A,
                                                      const __bf16* __restrict__ W,
                                                      __bf16* __restrict__ Cb,
                                                      float* __restrict__ Cf,
                                                      int M, int N, int K) {
    __shared__ __align__(16) __bf16 As[BM * TSTR];   // [m][k]
    __shared__ __align__(16) __bf16 Bs[BN * TSTR];   // [n][k]  (row-major W rows)

    const int tid  = threadIdx.x;
    const int lane = tid & 31;
    const int wave = tid >> 5;
    const int bm = blockIdx.y * BM;
    const int bn = blockIdx.x * BN;
    const int tileM0 = (wave & 3) * 32;
    const int tileN0 = (wave >> 2) * 64;

    const v8f vzero = {0.f, 0.f, 0.f, 0.f, 0.f, 0.f, 0.f, 0.f};
    v8f acc[2][4];
#pragma unroll
    for (int i = 0; i < 2; ++i)
#pragma unroll
        for (int j = 0; j < 4; ++j) acc[i][j] = vzero;

    const int cn  = lane & 15;
    const int kbA = (lane < 16) ? 0 : 16;
    const int kbB = (lane < 16) ? 0 : 16;
#if HAVE_TDM
    const unsigned int asoff = lds_offset_of(&As[0]);
    const unsigned int bsoff = lds_offset_of(&Bs[0]);
#endif

    for (int k0 = 0; k0 < K; k0 += BK) {
#if HAVE_TDM
        // TDM: 128 rows x 32 bf16, LDS rows padded 16+4 dwords (codes 3,3)
        if (wave == 0) {
            tdm_load_2d_bf16(A + (size_t)bm * K + k0, asoff, BK, BM, (unsigned)M, (unsigned)K, 3, 3);
            tdm_load_2d_bf16(W + (size_t)bn * K + k0, bsoff, BK, BM, (unsigned)N, (unsigned)K, 3, 3);
        }
#else
        for (int i = tid; i < BM * BK / 8; i += 256) {
            int r = i >> 2, sg = i & 3;
            async_b128(A + (size_t)(bm + r) * K + k0 + sg * 8, &As[r * TSTR + sg * 8]);
            async_b128(W + (size_t)(bn + r) * K + k0 + sg * 8, &Bs[r * TSTR + sg * 8]);
        }
#endif
        // prefetch next K tiles while this one is in flight
        if (k0 + BK < K) {
            int r = tid >> 1;
            __builtin_prefetch(A + (size_t)(bm + r) * K + k0 + BK, 0, 0);
            __builtin_prefetch(W + (size_t)(bn + r) * K + k0 + BK, 0, 0);
        }
#if HAVE_TDM
        if (wave == 0) __builtin_amdgcn_s_wait_tensorcnt(0);
#else
        async_wait_all();
#endif
        __syncthreads();

        v16bf afrag[2], bfrag[4];
#pragma unroll
        for (int t = 0; t < 2; ++t) {
            const __bf16* base = &As[(tileM0 + t * 16 + cn) * TSTR];
#pragma unroll
            for (int j = 0; j < 8; ++j) {
                afrag[t][j]     = base[kbA + j];
                afrag[t][8 + j] = base[kbA + 8 + j];
            }
        }
#pragma unroll
        for (int t = 0; t < 4; ++t) {
            const __bf16* base = &Bs[(tileN0 + t * 16 + cn) * TSTR + kbB];
#pragma unroll
            for (int e = 0; e < 16; ++e) bfrag[t][e] = base[e];
        }
#pragma unroll
        for (int i = 0; i < 2; ++i)
#pragma unroll
            for (int j = 0; j < 4; ++j)
                acc[i][j] = __builtin_amdgcn_wmma_f32_16x16x32_bf16(
                    false, afrag[i], false, bfrag[j], (short)0, acc[i][j], false, false);
        __syncthreads();
    }

    const int rbase = (lane < 16) ? 0 : 8;
#pragma unroll
    for (int i = 0; i < 2; ++i)
#pragma unroll
        for (int j = 0; j < 4; ++j)
#pragma unroll
            for (int r = 0; r < 8; ++r) {
                size_t grow = (size_t)(bm + tileM0 + i * 16 + rbase + r);
                size_t gcol = (size_t)(bn + tileN0 + j * 16 + cn);
                float v = acc[i][j][r];
                if (F32OUT) Cf[grow * N + gcol] = v;
                else        Cb[grow * N + gcol] = (__bf16)v;
            }
}

// ---------------------------------------------------------------- flash attention
#define FTQ 64
#define FNK 32
#define KSTR 72   // 64 data + 8 pad (bf16) per LDS row
#define PSTR 40

__global__ __launch_bounds__(128) void flash_attn_kernel(const __bf16* __restrict__ qkv,
                                                         const int* __restrict__ amask,
                                                         __bf16* __restrict__ out) {
    __shared__ __align__(16) __bf16 Ks[FNK * KSTR];   // [key][dh]
    __shared__ __align__(16) __bf16 Vs[FNK * KSTR];   // [key][dh]
    __shared__ __align__(16) __bf16 Ps[4][16 * PSTR];

    const int tid  = threadIdx.x;
    const int lane = tid & 31;
    const int wave = tid >> 5;
    const int b  = blockIdx.z;
    const int h  = blockIdx.y;
    const int q0 = blockIdx.x * FTQ;
    const int qr0 = q0 + wave * 16;

    const size_t rstride = 3 * Dc;
    const __bf16* qbase = qkv + (size_t)b * Tc * rstride + h * DHc;
    const __bf16* kbase = qbase + Dc;
    const __bf16* vbase = qbase + 2 * Dc;

    const int cn    = lane & 15;
    const int rbase = (lane < 16) ? 0 : 8;
    const int kbA   = (lane < 16) ? 0 : 16;
    const int kbB   = (lane < 16) ? 0 : 16;
#if HAVE_TDM
    const unsigned int ksoff = lds_offset_of(&Ks[0]);
    const unsigned int vsoff = lds_offset_of(&Vs[0]);
#endif

    // Q fragments (A layout), DH=64 -> 2 k-steps of 32
    v16bf qfrag[2];
    {
        const __bf16* qrow = qbase + (size_t)(qr0 + cn) * rstride;
#pragma unroll
        for (int t = 0; t < 2; ++t)
#pragma unroll
            for (int j = 0; j < 8; ++j) {
                qfrag[t][j]     = qrow[t * 32 + kbA + j];
                qfrag[t][8 + j] = qrow[t * 32 + kbA + 8 + j];
            }
    }

    const v8f vzero = {0.f, 0.f, 0.f, 0.f, 0.f, 0.f, 0.f, 0.f};
    v8f acc[4];
#pragma unroll
    for (int t = 0; t < 4; ++t) acc[t] = vzero;
    float mrow[8], lrow[8];
#pragma unroll
    for (int r = 0; r < 8; ++r) { mrow[r] = -3.0e38f; lrow[r] = 0.f; }

    const int smax = q0 + FTQ;
    for (int s0 = 0; s0 < smax; s0 += FNK) {
        // stage K & V tiles 32x64
#if HAVE_TDM
        // TDM: 32 rows x 64 bf16, LDS rows padded 32+4 dwords (codes 4,3)
        if (wave == 0) {
            tdm_load_2d_bf16(kbase + (size_t)s0 * rstride, ksoff, DHc, FNK,
                             (unsigned)Tc, (unsigned)rstride, 4, 3);
            tdm_load_2d_bf16(vbase + (size_t)s0 * rstride, vsoff, DHc, FNK,
                             (unsigned)Tc, (unsigned)rstride, 4, 3);
            __builtin_amdgcn_s_wait_tensorcnt(0);
        }
#else
        for (int i = tid; i < FNK * DHc / 8; i += 128) {
            int r = i >> 3, sg = i & 7;
            async_b128(kbase + (size_t)(s0 + r) * rstride + sg * 8, &Ks[r * KSTR + sg * 8]);
            async_b128(vbase + (size_t)(s0 + r) * rstride + sg * 8, &Vs[r * KSTR + sg * 8]);
        }
        async_wait_all();
#endif
        __syncthreads();

        // scores: Q(16x64) x K^T(64x32) -> two 16x16 tiles
        v8f sacc[2];
        sacc[0] = vzero; sacc[1] = vzero;
#pragma unroll
        for (int jt = 0; jt < 2; ++jt)
#pragma unroll
            for (int kk = 0; kk < 2; ++kk) {
                v16bf bfrag;
                const __bf16* p = &Ks[(jt * 16 + cn) * KSTR + kk * 32 + kbB];
#pragma unroll
                for (int e = 0; e < 16; ++e) bfrag[e] = p[e];
                sacc[jt] = __builtin_amdgcn_wmma_f32_16x16x32_bf16(
                    false, qfrag[kk], false, bfrag, (short)0, sacc[jt], false, false);
            }

        // scale + causal/pad mask
        float sc[2][8];
        int pad0 = amask[b * Tc + s0 + cn];
        int pad1 = amask[b * Tc + s0 + 16 + cn];
#pragma unroll
        for (int jt = 0; jt < 2; ++jt) {
            int col = s0 + jt * 16 + cn;
            int pad = jt ? pad1 : pad0;
#pragma unroll
            for (int r = 0; r < 8; ++r) {
                int grow = qr0 + rbase + r;
                bool ok = (col <= grow) && (pad != 0);
                sc[jt][r] = ok ? sacc[jt][r] * 0.125f : NEGINF;
            }
        }

        // online softmax (row reductions across 16 lanes of each half)
        float mnew[8], pscale[8];
#pragma unroll
        for (int r = 0; r < 8; ++r) {
            float v = fmaxf(sc[0][r], sc[1][r]);
            v = fmaxf(v, __shfl_xor(v, 1));
            v = fmaxf(v, __shfl_xor(v, 2));
            v = fmaxf(v, __shfl_xor(v, 4));
            v = fmaxf(v, __shfl_xor(v, 8));
            mnew[r]   = fmaxf(mrow[r], v);
            pscale[r] = __expf(mrow[r] - mnew[r]);
            mrow[r]   = mnew[r];
        }
        float pv[2][8];
#pragma unroll
        for (int r = 0; r < 8; ++r) {
            pv[0][r] = __expf(sc[0][r] - mnew[r]);
            pv[1][r] = __expf(sc[1][r] - mnew[r]);
            float srow = pv[0][r] + pv[1][r];
            srow += __shfl_xor(srow, 1);
            srow += __shfl_xor(srow, 2);
            srow += __shfl_xor(srow, 4);
            srow += __shfl_xor(srow, 8);
            lrow[r] = lrow[r] * pscale[r] + srow;
        }
#pragma unroll
        for (int t = 0; t < 4; ++t)
#pragma unroll
            for (int r = 0; r < 8; ++r) acc[t][r] *= pscale[r];

        // P: C-layout -> LDS -> A-layout
#pragma unroll
        for (int jt = 0; jt < 2; ++jt)
#pragma unroll
            for (int r = 0; r < 8; ++r)
                Ps[wave][(rbase + r) * PSTR + jt * 16 + cn] = (__bf16)pv[jt][r];
        __syncthreads();

        v16bf pfrag;
        {
            const __bf16* base = &Ps[wave][cn * PSTR];
#pragma unroll
            for (int j = 0; j < 8; ++j) {
                pfrag[j]     = base[kbA + j];
                pfrag[8 + j] = base[kbA + 8 + j];
            }
        }
#pragma unroll
        for (int t = 0; t < 4; ++t) {
            v16bf vfrag;
            const __bf16* base = &Vs[kbB * KSTR + t * 16 + cn];
#pragma unroll
            for (int e = 0; e < 16; ++e) vfrag[e] = base[e * KSTR];
            acc[t] = __builtin_amdgcn_wmma_f32_16x16x32_bf16(
                false, pfrag, false, vfrag, (short)0, acc[t], false, false);
        }
        __syncthreads();
    }

    // epilogue: out[b, t, h*64+dh] bf16
#pragma unroll
    for (int t = 0; t < 4; ++t)
#pragma unroll
        for (int r = 0; r < 8; ++r) {
            int grow = qr0 + rbase + r;
            float inv = (lrow[r] > 0.f) ? (1.0f / lrow[r]) : 0.f;
            out[((size_t)b * Tc + grow) * Dc + h * DHc + t * 16 + cn] =
                (__bf16)(acc[t][r] * inv);
        }
}

// ---------------------------------------------------------------- launch
extern "C" void kernel_launch(void* const* d_in, const int* in_sizes, int n_in,
                              void* d_out, int out_size, void* d_ws, size_t ws_size,
                              hipStream_t stream) {
    const float* x  = (const float*)d_in[0];
    const int* amask = (const int*)d_in[1];
    const float* Wq = (const float*)d_in[2];
    const float* Wk = (const float*)d_in[3];
    const float* Wv = (const float*)d_in[4];
    const float* Wo = (const float*)d_in[5];
    float* out = (float*)d_out;

    char* ws = (char*)d_ws;
    const size_t nX = (size_t)Bc * Tc * Dc;       // 8388608
    const size_t nW = (size_t)Dc * Dc;            // 1048576
    __bf16* x_bf  = (__bf16*)(ws);                 // 16 MB
    __bf16* wqkv  = (__bf16*)(ws + 16777216);      // 6 MB
    __bf16* wo_bf = (__bf16*)(ws + 23068672);      // 2 MB
    __bf16* qkv   = (__bf16*)(ws + 25165824);      // 48 MB
    __bf16* attn  = (__bf16*)(ws + 75497472);      // 16 MB
    float*  cosb  = (float*)(ws + 92274688);
    float*  sinb  = (float*)(ws + 92536832);

    f32_to_bf16_kernel<<<(nX + 255) / 256, 256, 0, stream>>>(x, x_bf, nX);
    f32_to_bf16_kernel<<<(nW + 255) / 256, 256, 0, stream>>>(Wq, wqkv, nW);
    f32_to_bf16_kernel<<<(nW + 255) / 256, 256, 0, stream>>>(Wk, wqkv + nW, nW);
    f32_to_bf16_kernel<<<(nW + 255) / 256, 256, 0, stream>>>(Wv, wqkv + 2 * nW, nW);
    f32_to_bf16_kernel<<<(nW + 255) / 256, 256, 0, stream>>>(Wo, wo_bf, nW);
    rope_tables_kernel<<<(Tc * 32 + 255) / 256, 256, 0, stream>>>(cosb, sinb);

    // fused QKV projection: [8192,1024] x [3072,1024]^T -> [8192,3072]
    gemm_bf16_wmma<false><<<dim3(3 * Dc / BN, Bc * Tc / BM), 256, 0, stream>>>(
        x_bf, wqkv, qkv, nullptr, Bc * Tc, 3 * Dc, Dc);

    rope_apply_kernel<<<((size_t)Bc * Tc * 2 * Hc * 32 + 255) / 256, 256, 0, stream>>>(
        qkv, cosb, sinb);

    flash_attn_kernel<<<dim3(Tc / FTQ, Hc, Bc), 128, 0, stream>>>(qkv, amask, attn);

    // output projection (f32 out): [8192,1024] x [1024,1024]^T
    gemm_bf16_wmma<true><<<dim3(Dc / BN, Bc * Tc / BM), 256, 0, stream>>>(
        attn, wo_bf, nullptr, out, Bc * Tc, Dc, Dc);
}
